// NonLocalBlock3D_33998961115395
// MI455X (gfx1250) — matrix operand order repaired and checked by reading, not compiled, not run on hardware
//
#include <hip/hip_runtime.h>

#define B_  4
#define C_  256
#define CI_ 128
#define N_  6272   // 8*28*28

typedef __attribute__((ext_vector_type(16))) _Float16 v16h;
typedef __attribute__((ext_vector_type(8)))  _Float16 v8h;
typedef __attribute__((ext_vector_type(8)))  float    v8f;

// DPP ROW_XMASK reduction step: max with lane (lane ^ MASK) within 16-lane row.
// Pure VALU (v_mov_b32_dpp + v_max) -- keeps the DS pipe free for the P bounce.
template <int MASK>
__device__ __forceinline__ float rowxor_maxf(float x) {
    int s = __builtin_amdgcn_mov_dpp(__float_as_int(x), 0x160 | MASK, 0xf, 0xf, true);
    return fmaxf(x, __int_as_float(s));
}

// ---------------------------------------------------------------------------
// Kernel 1: projections th = Wt@x+bt, phT = Wp@x+bp, gT = Wg@x+bg  (f16 out)
// th/phT stored [B,N,CI] (row n contiguous over o), gT stored [B,CI,N].
// ---------------------------------------------------------------------------
__global__ __launch_bounds__(256)
void proj_kernel(const float* __restrict__ x,
                 const float* __restrict__ Wg, const float* __restrict__ bg,
                 const float* __restrict__ Wt, const float* __restrict__ bt,
                 const float* __restrict__ Wp, const float* __restrict__ bp,
                 _Float16* __restrict__ th, _Float16* __restrict__ phT,
                 _Float16* __restrict__ gT)
{
    int idx  = blockIdx.x * blockDim.x + threadIdx.x;
    int o    = idx & (CI_ - 1);
    int rest = idx >> 7;
    int n    = rest % N_;
    int b    = rest / N_;
    if (b >= B_) return;

    const float* xb = x + ((size_t)b * C_) * N_ + n;   // x[b,c,n] = xb[c*N_]
    const float* wt = Wt + (size_t)o * C_;
    const float* wp = Wp + (size_t)o * C_;
    const float* wg = Wg + (size_t)o * C_;

    float st = bt[o], sp = bp[o], sg = bg[o];
    #pragma unroll 8
    for (int c = 0; c < C_; ++c) {
        float xv = xb[(size_t)c * N_];   // wave-uniform broadcast read
        st = fmaf(wt[c], xv, st);
        sp = fmaf(wp[c], xv, sp);
        sg = fmaf(wg[c], xv, sg);
    }
    size_t no = ((size_t)b * N_ + n) * CI_ + o;
    th[no]  = (_Float16)st;
    phT[no] = (_Float16)sp;
    gT[((size_t)b * CI_ + o) * N_ + n] = (_Float16)sg;
}

// ---------------------------------------------------------------------------
// Kernel 2: flash-attention core with v_wmma_f32_16x16x32_f16.
// One wave handles a 16-row n-strip, streaming m in chunks of 32 with online
// softmax. Softmax denominator accumulated by a WMMA against a ones-matrix
// (same D-layout as y), row-max reduced with DPP row_xmask.
// y (pre-output-projection) written as f32 [B,N,CI].
// ---------------------------------------------------------------------------
__global__ __launch_bounds__(256)
void attn_kernel(const _Float16* __restrict__ th,
                 const _Float16* __restrict__ phT,
                 const _Float16* __restrict__ gT,
                 float* __restrict__ y)
{
    __shared__ _Float16 sP[8][16 * 32];     // per-wave P tile (16n x 32m), 8 KB
    const int lane = threadIdx.x & 31;
    const int wave = threadIdx.x >> 5;
    const int h    = lane >> 4;             // 16-lane half id
    const int lr   = lane & 15;
    const int b    = blockIdx.y;
    const int n0   = blockIdx.x * 128 + wave * 16;

    const _Float16* thB = th  + (size_t)b * N_ * CI_;
    const _Float16* phB = phT + (size_t)b * N_ * CI_;
    const _Float16* gB  = gT  + (size_t)b * CI_ * N_;
    _Float16* sPw = sP[wave];

    // A fragments of th (16n x 32c per chunk), resident for the whole m loop.
    // A layout per lane: K = [16h,16h+8) in VGPR0-3, [16h+8,16h+16) in VGPR4-7.
    v16h aT[4];
    #pragma unroll
    for (int kc = 0; kc < 4; ++kc) {
        const _Float16* rp = thB + (size_t)(n0 + lr) * CI_ + kc * 32;
        v8h lo = *(const v8h*)(rp + 16 * h);
        v8h hi = *(const v8h*)(rp + 16 * h + 8);
        aT[kc] = __builtin_shufflevector(lo, hi,
                   0,1,2,3,4,5,6,7,8,9,10,11,12,13,14,15);
    }

    // B fragment of all-ones: row-sum generator for the softmax denominator.
    v16h onesv;
    #pragma unroll
    for (int i = 0; i < 16; ++i) onesv[i] = (_Float16)1.0f;

    v8f zero = {};
    v8f acc[8];
    #pragma unroll
    for (int ot = 0; ot < 8; ++ot) acc[ot] = zero;
    v8f accL = zero;                        // running softmax denominator

    float mrun[8];
    #pragma unroll
    for (int v = 0; v < 8; ++v) mrun[v] = -1e30f;

    for (int m0 = 0; m0 < N_; m0 += 32) {
        // Keep the 8 waves of this block in lockstep: they read identical
        // phT/gT lines this iteration, so waves 2..8 hit WGP$/L1.
        __syncthreads();

        // ---- S = th . ph^T for this 16x32 block (two 16x16 D tiles) ----
        v8f f0 = zero, f1 = zero;
        #pragma unroll
        for (int kc = 0; kc < 4; ++kc) {
            v16h b0 = *(const v16h*)(phB + (size_t)(m0      + lr) * CI_ + kc * 32 + 16 * h);
            v16h b1 = *(const v16h*)(phB + (size_t)(m0 + 16 + lr) * CI_ + kc * 32 + 16 * h);
            f0 = __builtin_amdgcn_wmma_f32_16x16x32_f16(false, aT[kc], false, b0,
                                                        (short)0, f0, false, false);
            f1 = __builtin_amdgcn_wmma_f32_16x16x32_f16(false, aT[kc], false, b1,
                                                        (short)0, f1, false, false);
        }

        // ---- online softmax: row max via DPP row_xmask (row r = v + 8h) ----
        float sc[8];
        #pragma unroll
        for (int v = 0; v < 8; ++v) {
            float a = fmaxf(f0[v], f1[v]);
            a = rowxor_maxf<1>(a);
            a = rowxor_maxf<2>(a);
            a = rowxor_maxf<4>(a);
            a = rowxor_maxf<8>(a);
            float newm = fmaxf(mrun[v], a);
            sc[v]   = __expf(mrun[v] - newm);
            mrun[v] = newm;
            float p0 = __expf(f0[v] - newm);
            float p1 = __expf(f1[v] - newm);
            sPw[(v + 8 * h) * 32 + lr]      = (_Float16)p0;
            sPw[(v + 8 * h) * 32 + 16 + lr] = (_Float16)p1;
        }

        // ---- rescale running accumulators (y tiles + denominator) ----
        #pragma unroll
        for (int v = 0; v < 8; ++v) accL[v] *= sc[v];
        #pragma unroll
        for (int ot = 0; ot < 8; ++ot)
            #pragma unroll
            for (int v = 0; v < 8; ++v)
                acc[ot][v] *= sc[v];

        // ---- re-layout P (D-layout -> A-fragment) via per-wave LDS ----
        asm volatile("s_wait_dscnt 0" ::: "memory");
        v8h plo = *(const v8h*)(sPw + lr * 32 + 16 * h);
        v8h phi = *(const v8h*)(sPw + lr * 32 + 16 * h + 8);
        v16h pa = __builtin_shufflevector(plo, phi,
                    0,1,2,3,4,5,6,7,8,9,10,11,12,13,14,15);

        // ---- denominator += P . ones ; y += P . g (8 o-tiles) ----
        accL = __builtin_amdgcn_wmma_f32_16x16x32_f16(false, pa, false, onesv,
                                                      (short)0, accL, false, false);
        #pragma unroll
        for (int ot = 0; ot < 8; ++ot) {
            v16h gb = *(const v16h*)(gB + (size_t)(ot * 16 + lr) * N_ + m0 + 16 * h);
            acc[ot] = __builtin_amdgcn_wmma_f32_16x16x32_f16(false, pa, false, gb,
                                                             (short)0, acc[ot], false, false);
        }
    }

    // ---- normalize (v_rcp + mul, not fdiv) and store y[b,n,o] ----
    float invl[8];
    #pragma unroll
    for (int v = 0; v < 8; ++v) invl[v] = __builtin_amdgcn_rcpf(accL[v]);

    #pragma unroll
    for (int ot = 0; ot < 8; ++ot) {
        #pragma unroll
        for (int v = 0; v < 8; ++v) {
            int n = n0 + v + 8 * h;
            int o = ot * 16 + lr;
            y[((size_t)b * N_ + n) * CI_ + o] = acc[ot][v] * invl[v];
        }
    }
}

// ---------------------------------------------------------------------------
// Kernel 3: wy = Ww@y + bw, eval BatchNorm, residual add, f32 out [B,C,N].
// ---------------------------------------------------------------------------
__global__ __launch_bounds__(256)
void out_kernel(const float* __restrict__ y, const float* __restrict__ x,
                const float* __restrict__ Ww, const float* __restrict__ bw,
                const float* __restrict__ gamma, const float* __restrict__ beta,
                const float* __restrict__ rmean, const float* __restrict__ rvar,
                float* __restrict__ out)
{
    int idx  = blockIdx.x * blockDim.x + threadIdx.x;
    int n    = idx % N_;
    int rest = idx / N_;
    int c    = rest % C_;
    int b    = rest / C_;
    if (b >= B_) return;

    const float* yb = y + ((size_t)b * N_ + n) * CI_;
    const float* wc = Ww + (size_t)c * CI_;
    float acc = 0.f;
    #pragma unroll 8
    for (int o = 0; o < CI_; ++o) acc = fmaf(wc[o], yb[o], acc);

    float inv = gamma[c] * rsqrtf(rvar[c] + 1e-5f);
    size_t xi = ((size_t)b * C_ + c) * N_ + n;
    out[xi] = (acc + bw[c]) * inv + (beta[c] - rmean[c] * inv) + x[xi];
}

// ---------------------------------------------------------------------------
extern "C" void kernel_launch(void* const* d_in, const int* in_sizes, int n_in,
                              void* d_out, int out_size, void* d_ws, size_t ws_size,
                              hipStream_t stream)
{
    const float* x     = (const float*)d_in[0];
    const float* Wg    = (const float*)d_in[1];
    const float* bg    = (const float*)d_in[2];
    const float* Wt    = (const float*)d_in[3];
    const float* bt    = (const float*)d_in[4];
    const float* Wp    = (const float*)d_in[5];
    const float* bp    = (const float*)d_in[6];
    const float* Ww    = (const float*)d_in[7];
    const float* bw    = (const float*)d_in[8];
    const float* gamma = (const float*)d_in[9];
    const float* beta  = (const float*)d_in[10];
    const float* rmean = (const float*)d_in[11];
    const float* rvar  = (const float*)d_in[12];
    float* out = (float*)d_out;

    char* ws = (char*)d_ws;
    const size_t projElems = (size_t)B_ * N_ * CI_;
    _Float16* th  = (_Float16*)ws; ws += projElems * sizeof(_Float16);
    _Float16* phT = (_Float16*)ws; ws += projElems * sizeof(_Float16);
    _Float16* gT  = (_Float16*)ws; ws += projElems * sizeof(_Float16);
    float*    y   = (float*)ws;    ws += projElems * sizeof(float);

    {   // projections: B*N*CI threads
        int total = B_ * N_ * CI_;
        proj_kernel<<<dim3(total / 256), dim3(256), 0, stream>>>(
            x, Wg, bg, Wt, bt, Wp, bp, th, phT, gT);
    }
    {   // flash attention: 8 waves/block, each a 16-row strip
        dim3 grid(N_ / 128, B_);
        attn_kernel<<<grid, dim3(256), 0, stream>>>(th, phT, gT, y);
    }
    {   // output projection + BN + residual
        int total = B_ * C_ * N_;
        out_kernel<<<dim3(total / 256), dim3(256), 0, stream>>>(
            y, x, Ww, bw, gamma, beta, rmean, rvar, out);
    }
}
